// EnhancedAttention_62723702391371
// MI455X (gfx1250) — matrix operand order repaired and checked by reading, not compiled, not run on hardware
//
#include <hip/hip_runtime.h>
#include <hip/hip_bf16.h>
#include <math.h>

typedef __attribute__((ext_vector_type(16))) _Float16 v16h;
typedef __attribute__((ext_vector_type(8)))  float    v8f;
typedef __attribute__((ext_vector_type(4)))  float    v4f;

#define DEVI __device__ __forceinline__

constexpr int B_ = 32, C_ = 256, H_ = 64, W_ = 64, MID = 16;
constexpr int HW  = H_ * W_;
constexpr int QKV = 3 * MID;   // 48

DEVI float gelu_f(float x) { return 0.5f * x * (1.0f + erff(x * 0.70710678118654752f)); }
DEVI float sigm_f(float x) { return 1.0f / (1.0f + expf(-x)); }

// CDNA5 WMMA 16x16x32 f16 fragment index maps (ISA 7.12.2, wave32).
// A (16x32, MxK): lane L holds row M=L%16; half i -> K = (i<8 ? i : i+8) + (L>=16 ? 8 : 0)
// B (32x16, KxN): lane L holds col N=L%16; half i -> K = (i<8 ? i : i+8) + (L>=16 ? 8 : 0)
// C/D (16x16):    lane L holds col N=L%16; float r -> M = (L>=16 ? 8 : 0) + r
DEVI int kmapA(int i, int hi) { return (i < 8 ? i : i + 8) + (hi ? 8 : 0); }
DEVI int kmapB(int i, int hi) { return (i < 8 ? i : i + 8) + (hi ? 8 : 0); }

DEVI v8f wmma_f16(v16h a, v16h b, v8f c) {
  return __builtin_amdgcn_wmma_f32_16x16x32_f16(false, a, false, b, (short)0, c, false, false);
}

// ---------------------------------------------------------------- K1: gs = mean_hw(x)
__global__ __launch_bounds__(256) void k_reduce_gs(const float* __restrict__ x,
                                                   float* __restrict__ gs) {
  const int bc = blockIdx.x;
  const v4f* p4 = (const v4f*)(x + (size_t)bc * HW);
  float s = 0.f;
  for (int i = threadIdx.x; i < HW / 4; i += 256) {
    __builtin_prefetch(p4 + i + HW / 4, 0, 1);   // speculative: next plane
    v4f v = p4[i];
    s += v.x + v.y + v.z + v.w;
  }
  __shared__ float red[256];
  red[threadIdx.x] = s;
  __syncthreads();
  for (int t = 128; t > 0; t >>= 1) {
    if (threadIdx.x < t) red[threadIdx.x] += red[threadIdx.x + t];
    __syncthreads();
  }
  if (threadIdx.x == 0) gs[bc] = red[0] * (1.0f / HW);
}

// ---------------------------------------------------------------- K2: global MLP -> gw
__global__ __launch_bounds__(512) void k_global_mlp(const float* __restrict__ gs,
                                                    const float* __restrict__ fc1_w,
                                                    const float* __restrict__ fc1_b,
                                                    const float* __restrict__ fc2_w,
                                                    const float* __restrict__ fc2_b,
                                                    float* __restrict__ gw) {
  __shared__ float gsl[B_ * C_];
  __shared__ float hl[B_ * MID];
  for (int i = threadIdx.x; i < B_ * C_; i += 512) gsl[i] = gs[i];
  __syncthreads();
  if (threadIdx.x < B_ * MID) {
    int b = threadIdx.x / MID, m = threadIdx.x % MID;
    float s = fc1_b[m];
    for (int c = 0; c < C_; ++c) s += gsl[b * C_ + c] * fc1_w[m * C_ + c];
    hl[threadIdx.x] = fmaxf(s, 0.f);
  }
  __syncthreads();
  for (int i = threadIdx.x; i < B_ * C_; i += 512) {
    int b = i / C_, c = i % C_;
    float s = fc2_b[c];
    for (int m = 0; m < MID; ++m) s += hl[b * MID + m] * fc2_w[c * MID + m];
    gw[i] = sigm_f(s);
  }
}

// ---------------------------------------------------------------- K3: f32->f16 weight convert
__global__ __launch_bounds__(256) void k_cvt_w(const float* __restrict__ row_w,
                                               const float* __restrict__ col_w,
                                               const float* __restrict__ ax_w,
                                               _Float16* __restrict__ row_wh,
                                               _Float16* __restrict__ col_wh,
                                               _Float16* __restrict__ ax_wh) {
  for (int i = threadIdx.x; i < QKV * C_; i += 256) row_wh[i] = (_Float16)row_w[i];
  for (int i = threadIdx.x; i < QKV * MID; i += 256) col_wh[i] = (_Float16)col_w[i];
  for (int i = threadIdx.x; i < C_ * MID; i += 256) ax_wh[i] = (_Float16)ax_w[i];
}

// ---------------------------------------------------------------- K4: local branch -> zsum[b][16]
// z = gelu(pw1(gelu(dw2(gelu(dw1(x)))))); pw2 is linear so mean(pw2(z)) = pw2(mean(z)).
__global__ __launch_bounds__(256) void k_local(const float* __restrict__ x,
                                               const float* __restrict__ dw1_w,
                                               const float* __restrict__ dw1_b,
                                               const float* __restrict__ dw2_w,
                                               const float* __restrict__ dw2_b,
                                               const float* __restrict__ pw1_w,
                                               const float* __restrict__ pw1_b,
                                               float* __restrict__ zsum) {
  const int tile = blockIdx.x;      // 16 row tiles of 4 rows
  const int b = blockIdx.y;
  const int h0 = tile * 4;
  const int tid = threadIdx.x;
  const int r = tid >> 6, w = tid & 63;   // pixel owned by this thread
  __shared__ float xrow[6 * 64];
  __shared__ float t1s[6 * 64];
  __shared__ float pw1l[MID * C_];
  __shared__ float zp[MID];
  for (int i = tid; i < MID * C_; i += 256) pw1l[i] = pw1_w[i];
  if (tid < MID) zp[tid] = 0.f;
  float acc[MID];
#pragma unroll
  for (int m = 0; m < MID; ++m) acc[m] = 0.f;

  for (int c = 0; c < C_; ++c) {
    const float a0 = dw1_w[c * 3 + 0], a1 = dw1_w[c * 3 + 1], a2 = dw1_w[c * 3 + 2], ab = dw1_b[c];
    const float v0 = dw2_w[c * 3 + 0], v1 = dw2_w[c * 3 + 1], v2 = dw2_w[c * 3 + 2], vb = dw2_b[c];
    __syncthreads();   // previous-iteration t1s reads done before restaging
    for (int i = tid; i < 384; i += 256) {
      int rr = i >> 6, ww = i & 63, gh = h0 - 1 + rr;
      xrow[i] = (gh >= 0 && gh < H_) ? x[(((size_t)b * C_ + c) * H_ + gh) * W_ + ww] : 0.f;
    }
    __syncthreads();
    for (int i = tid; i < 384; i += 256) {
      int rr = i >> 6, ww = i & 63, gh = h0 - 1 + rr;
      float t = 0.f;
      if (gh >= 0 && gh < H_) {        // rows outside H are dw2's zero padding (post-gelu)
        float xm = (ww > 0) ? xrow[rr * 64 + ww - 1] : 0.f;
        float xc = xrow[rr * 64 + ww];
        float xp = (ww < 63) ? xrow[rr * 64 + ww + 1] : 0.f;
        t = gelu_f(ab + a0 * xm + a1 * xc + a2 * xp);
      }
      t1s[i] = t;
    }
    __syncthreads();
    float t2 = gelu_f(vb + v0 * t1s[(r + 0) * 64 + w] + v1 * t1s[(r + 1) * 64 + w] +
                      v2 * t1s[(r + 2) * 64 + w]);
#pragma unroll
    for (int m = 0; m < MID; ++m) acc[m] += pw1l[m * C_ + c] * t2;
  }
  __syncthreads();
#pragma unroll
  for (int m = 0; m < MID; ++m) {
    float z = gelu_f(acc[m] + pw1_b[m]);
    z += __shfl_xor(z, 1);  z += __shfl_xor(z, 2);  z += __shfl_xor(z, 4);
    z += __shfl_xor(z, 8);  z += __shfl_xor(z, 16);
    if ((tid & 31) == 0) atomicAdd(&zp[m], z);
  }
  __syncthreads();
  if (tid < MID) atomicAdd(&zsum[b * MID + tid], zp[tid]);
}

// ---------------------------------------------------------------- K5: row attention per (b,h)
__global__ __launch_bounds__(256) void k_rowattn(const float* __restrict__ x,
                                                 const _Float16* __restrict__ row_wh,
                                                 const float* __restrict__ row_b,
                                                 float* __restrict__ orow) {
  extern __shared__ char smem[];
  constexpr int XSTR = C_ + 8;                           // 264 halves/row
  _Float16* xs  = (_Float16*)smem;                       // [64][XSTR]  f16
  float*    qkv = (float*)(smem + 64 * XSTR * 2);        // [64][50]    f32
  float*    sl  = qkv + 64 * 50;                         // [64][66]    f32 logits
  _Float16* pm  = (_Float16*)(sl + 64 * 66);             // [64][64]    f16 probs

  const int h = blockIdx.x, b = blockIdx.y;
  const int tid = threadIdx.x, lane = tid & 31, wave = tid >> 5;
  const int l15 = lane & 15, hi = lane >> 4;

  // Stage x[b,:,h,:] transposed to [w][c] as f16 (vectorized b128 global reads).
  for (int i = tid; i < C_ * (W_ / 4); i += 256) {
    int c = i >> 4, w4 = (i & 15) << 2;
    v4f v = *(const v4f*)(x + (((size_t)b * C_ + c) * H_ + h) * W_ + w4);
    xs[(w4 + 0) * XSTR + c] = (_Float16)v.x;
    xs[(w4 + 1) * XSTR + c] = (_Float16)v.y;
    xs[(w4 + 2) * XSTR + c] = (_Float16)v.z;
    xs[(w4 + 3) * XSTR + c] = (_Float16)v.w;
  }
  __syncthreads();

  // QKV = xs(64x256) @ row_w^T(256x48) + row_b : 4 M-tiles x 3 N-tiles, K=256 (8 WMMA steps)
  for (int t = wave; t < 12; t += 8) {
    int mt = t / 3, nt = t % 3;
    v8f acc = {};
    for (int ks = 0; ks < 8; ++ks) {
      v16h a = {}, bb = {};
#pragma unroll
      for (int i = 0; i < 16; ++i) {
        a[i]  = xs[(mt * 16 + l15) * XSTR + ks * 32 + kmapA(i, hi)];
        bb[i] = row_wh[(nt * 16 + l15) * C_ + ks * 32 + kmapB(i, hi)];
      }
      acc = wmma_f16(a, bb, acc);
    }
    float bias = row_b[nt * 16 + l15];
#pragma unroll
    for (int rr = 0; rr < 8; ++rr)
      qkv[(mt * 16 + hi * 8 + rr) * 50 + nt * 16 + l15] = acc[rr] + bias;
  }
  __syncthreads();

  // S = q @ k^T * (1/sqrt(16)) : 4x4 tiles, K=16 zero-padded to 32
  for (int t = wave; t < 16; t += 8) {
    int mt = t >> 2, nt = t & 3;
    v16h a = {}, bb = {};
#pragma unroll
    for (int i = 0; i < 16; ++i) {
      int ka = kmapA(i, hi);
      a[i] = (ka < MID) ? (_Float16)qkv[(mt * 16 + l15) * 50 + ka] : (_Float16)0.f;
      int kb = kmapB(i, hi);
      bb[i] = (kb < MID) ? (_Float16)qkv[(nt * 16 + l15) * 50 + 16 + kb] : (_Float16)0.f;
    }
    v8f acc = {};
    acc = wmma_f16(a, bb, acc);
#pragma unroll
    for (int rr = 0; rr < 8; ++rr)
      sl[(mt * 16 + hi * 8 + rr) * 66 + nt * 16 + l15] = acc[rr] * 0.25f;
  }
  __syncthreads();

  // f32 softmax per query row
  if (tid < 64) {
    float mx = -1e30f;
    for (int j = 0; j < 64; ++j) mx = fmaxf(mx, sl[tid * 66 + j]);
    float s = 0.f;
    for (int j = 0; j < 64; ++j) { float e = expf(sl[tid * 66 + j] - mx); sl[tid * 66 + j] = e; s += e; }
    float inv = 1.0f / s;
    for (int j = 0; j < 64; ++j) pm[tid * 64 + j] = (_Float16)(sl[tid * 66 + j] * inv);
  }
  __syncthreads();

  // out_row = P(64x64) @ V(64x16): 4 M-tiles, K=64 (2 WMMA steps)
  if (wave < 4) {
    int mt = wave;
    v8f acc = {};
    for (int ks = 0; ks < 2; ++ks) {
      v16h a = {}, bb = {};
#pragma unroll
      for (int i = 0; i < 16; ++i) {
        a[i]  = pm[(mt * 16 + l15) * 64 + ks * 32 + kmapA(i, hi)];
        bb[i] = (_Float16)qkv[(ks * 32 + kmapB(i, hi)) * 50 + 32 + l15];
      }
      acc = wmma_f16(a, bb, acc);
    }
#pragma unroll
    for (int rr = 0; rr < 8; ++rr) {
      int w = mt * 16 + hi * 8 + rr;
      orow[(((size_t)b * W_ + w) * H_ + h) * MID + l15] = acc[rr];   // layout [b][w][h][mid]
    }
  }
}

// ---------------------------------------------------------------- K6: col attention + ax per (b,w)
__global__ __launch_bounds__(256) void k_colattn(const float* __restrict__ orow,
                                                 const _Float16* __restrict__ col_wh,
                                                 const float* __restrict__ col_b,
                                                 const _Float16* __restrict__ ax_wh,
                                                 const float* __restrict__ ax_b,
                                                 float* __restrict__ axsum) {
  extern __shared__ char smem[];
  _Float16* xc  = (_Float16*)smem;                       // [64][18] f16
  float*    qkv = (float*)(smem + 64 * 18 * 2);          // [64][50]
  float*    sl  = qkv + 64 * 50;                         // [64][66]
  _Float16* pm  = (_Float16*)(sl + 64 * 66);             // [64][64]
  _Float16* ocl = pm + 64 * 64;                          // [64][18]
  float*    axp = (float*)(ocl + 64 * 18);               // [256]

  const int w = blockIdx.x, b = blockIdx.y;
  const int tid = threadIdx.x, lane = tid & 31, wave = tid >> 5;
  const int l15 = lane & 15, hi = lane >> 4;

  {
    // contiguous 64x16 f32 slab -> one b128 per thread
    v4f v = ((const v4f*)(orow + (((size_t)b * W_ + w) * H_) * MID))[tid];
    int hrow = tid >> 2, d = (tid & 3) << 2;
    xc[hrow * 18 + d + 0] = (_Float16)v.x;
    xc[hrow * 18 + d + 1] = (_Float16)v.y;
    xc[hrow * 18 + d + 2] = (_Float16)v.z;
    xc[hrow * 18 + d + 3] = (_Float16)v.w;
  }
  axp[tid] = 0.f;
  __syncthreads();

  // qkv2 = xc(64x16) @ col_w^T(16x48) + col_b : K=16 padded to 32
  for (int t = wave; t < 12; t += 8) {
    int mt = t / 3, nt = t % 3;
    v16h a = {}, bb = {};
#pragma unroll
    for (int i = 0; i < 16; ++i) {
      int ka = kmapA(i, hi);
      a[i] = (ka < MID) ? xc[(mt * 16 + l15) * 18 + ka] : (_Float16)0.f;
      int kb = kmapB(i, hi);
      bb[i] = (kb < MID) ? col_wh[(nt * 16 + l15) * MID + kb] : (_Float16)0.f;
    }
    v8f acc = {};
    acc = wmma_f16(a, bb, acc);
    float bias = col_b[nt * 16 + l15];
#pragma unroll
    for (int rr = 0; rr < 8; ++rr)
      qkv[(mt * 16 + hi * 8 + rr) * 50 + nt * 16 + l15] = acc[rr] + bias;
  }
  __syncthreads();

  for (int t = wave; t < 16; t += 8) {
    int mt = t >> 2, nt = t & 3;
    v16h a = {}, bb = {};
#pragma unroll
    for (int i = 0; i < 16; ++i) {
      int ka = kmapA(i, hi);
      a[i] = (ka < MID) ? (_Float16)qkv[(mt * 16 + l15) * 50 + ka] : (_Float16)0.f;
      int kb = kmapB(i, hi);
      bb[i] = (kb < MID) ? (_Float16)qkv[(nt * 16 + l15) * 50 + 16 + kb] : (_Float16)0.f;
    }
    v8f acc = {};
    acc = wmma_f16(a, bb, acc);
#pragma unroll
    for (int rr = 0; rr < 8; ++rr)
      sl[(mt * 16 + hi * 8 + rr) * 66 + nt * 16 + l15] = acc[rr] * 0.25f;
  }
  __syncthreads();

  if (tid < 64) {
    float mx = -1e30f;
    for (int j = 0; j < 64; ++j) mx = fmaxf(mx, sl[tid * 66 + j]);
    float s = 0.f;
    for (int j = 0; j < 64; ++j) { float e = expf(sl[tid * 66 + j] - mx); sl[tid * 66 + j] = e; s += e; }
    float inv = 1.0f / s;
    for (int j = 0; j < 64; ++j) pm[tid * 64 + j] = (_Float16)(sl[tid * 66 + j] * inv);
  }
  __syncthreads();

  if (wave < 4) {
    int mt = wave;
    v8f acc = {};
    for (int ks = 0; ks < 2; ++ks) {
      v16h a = {}, bb = {};
#pragma unroll
      for (int i = 0; i < 16; ++i) {
        a[i]  = pm[(mt * 16 + l15) * 64 + ks * 32 + kmapA(i, hi)];
        bb[i] = (_Float16)qkv[(ks * 32 + kmapB(i, hi)) * 50 + 32 + l15];
      }
      acc = wmma_f16(a, bb, acc);
    }
#pragma unroll
    for (int rr = 0; rr < 8; ++rr)
      ocl[(mt * 16 + hi * 8 + rr) * 18 + l15] = (_Float16)acc[rr];
  }
  __syncthreads();

  // ax = sigmoid(out_col(64x16) @ ax_w^T(16x256) + ax_b); sum over the 64 h rows
  for (int t = wave; t < 64; t += 8) {
    int mt = t >> 4, nt = t & 15;
    v16h a = {}, bb = {};
#pragma unroll
    for (int i = 0; i < 16; ++i) {
      int ka = kmapA(i, hi);
      a[i] = (ka < MID) ? ocl[(mt * 16 + l15) * 18 + ka] : (_Float16)0.f;
      int kb = kmapB(i, hi);
      bb[i] = (kb < MID) ? ax_wh[(nt * 16 + l15) * MID + kb] : (_Float16)0.f;
    }
    v8f acc = {};
    acc = wmma_f16(a, bb, acc);
    float bias = ax_b[nt * 16 + l15];
    float ssum = 0.f;
#pragma unroll
    for (int rr = 0; rr < 8; ++rr) ssum += sigm_f(acc[rr] + bias);
    atomicAdd(&axp[nt * 16 + l15], ssum);
  }
  __syncthreads();
  atomicAdd(&axsum[b * C_ + tid], axp[tid]);
}

// ---------------------------------------------------------------- K7: fuse gates
__global__ __launch_bounds__(256) void k_fuse(const float* __restrict__ gw,
                                              const float* __restrict__ zsum,
                                              const float* __restrict__ axsum,
                                              const float* __restrict__ pw2_w,
                                              const float* __restrict__ pw2_b,
                                              const float* __restrict__ fusion_w,
                                              float* __restrict__ finalw) {
  float f0 = fusion_w[0], f1 = fusion_w[1], f2 = fusion_w[2], f3 = fusion_w[3];
  float mx = fmaxf(fmaxf(f0, f1), fmaxf(f2, f3));
  float e0 = expf(f0 - mx), e1 = expf(f1 - mx), e2 = expf(f2 - mx), e3 = expf(f3 - mx);
  float inv = 1.0f / (e0 + e1 + e2 + e3);
  e0 *= inv; e1 *= inv; e2 *= inv; e3 *= inv;
  for (int i = threadIdx.x; i < B_ * C_; i += 256) {
    int b = i / C_, c = i % C_;
    float lw = pw2_b[c];
    for (int m = 0; m < MID; ++m) lw += pw2_w[c * MID + m] * (zsum[b * MID + m] * (1.0f / HW));
    lw = sigm_f(lw);
    float axw = axsum[i] * (1.0f / HW);
    finalw[i] = e0 * gw[i] + e1 * lw + e2 * axw + e3;
  }
}

// ---------------------------------------------------------------- K8: y = x * final (streaming)
// NT stores keep the write-once output from evicting L2-resident x.
__global__ __launch_bounds__(256) void k_apply(const v4f* __restrict__ xv,
                                               const float* __restrict__ finalw,
                                               v4f* __restrict__ yv, int n4) {
  int i = blockIdx.x * 256 + threadIdx.x;
  if (i >= n4) return;
  float f = finalw[i >> 10];   // 1024 float4 per (b,c) plane
  v4f v = xv[i] * f;
  __builtin_nontemporal_store(v, yv + i);
}

// ----------------------------------------------------------------
extern "C" void kernel_launch(void* const* d_in, const int* in_sizes, int n_in,
                              void* d_out, int out_size, void* d_ws, size_t ws_size,
                              hipStream_t stream) {
  const float* x     = (const float*)d_in[0];
  const float* fc1_w = (const float*)d_in[1];  const float* fc1_b = (const float*)d_in[2];
  const float* fc2_w = (const float*)d_in[3];  const float* fc2_b = (const float*)d_in[4];
  const float* dw1_w = (const float*)d_in[5];  const float* dw1_b = (const float*)d_in[6];
  const float* dw2_w = (const float*)d_in[7];  const float* dw2_b = (const float*)d_in[8];
  const float* pw1_w = (const float*)d_in[9];  const float* pw1_b = (const float*)d_in[10];
  const float* pw2_w = (const float*)d_in[11]; const float* pw2_b = (const float*)d_in[12];
  const float* row_w = (const float*)d_in[13]; const float* row_b = (const float*)d_in[14];
  const float* col_w = (const float*)d_in[15]; const float* col_b = (const float*)d_in[16];
  const float* ax_w  = (const float*)d_in[17]; const float* ax_b  = (const float*)d_in[18];
  const float* fusion_w = (const float*)d_in[19];
  float* out = (float*)d_out;
  (void)in_sizes; (void)n_in; (void)out_size; (void)ws_size;

  char* p = (char*)d_ws;
  auto alloc = [&](size_t bytes) -> char* {
    char* r = p; p += (bytes + 255) & ~(size_t)255; return r;
  };
  float*    gs     = (float*)alloc((size_t)B_ * C_ * 4);
  float*    gw     = (float*)alloc((size_t)B_ * C_ * 4);
  float*    zsum   = (float*)alloc((size_t)B_ * MID * 4);
  float*    axsum  = (float*)alloc((size_t)B_ * C_ * 4);
  float*    finalw = (float*)alloc((size_t)B_ * C_ * 4);
  float*    orow   = (float*)alloc((size_t)B_ * W_ * H_ * MID * 4);   // 8 MB
  _Float16* row_wh = (_Float16*)alloc((size_t)QKV * C_ * 2);
  _Float16* col_wh = (_Float16*)alloc((size_t)QKV * MID * 2);
  _Float16* ax_wh  = (_Float16*)alloc((size_t)C_ * MID * 2);

  hipMemsetAsync(zsum, 0, (size_t)B_ * MID * 4, stream);
  hipMemsetAsync(axsum, 0, (size_t)B_ * C_ * 4, stream);

  k_reduce_gs<<<B_ * C_, 256, 0, stream>>>(x, gs);
  k_global_mlp<<<1, 512, 0, stream>>>(gs, fc1_w, fc1_b, fc2_w, fc2_b, gw);
  k_cvt_w<<<1, 256, 0, stream>>>(row_w, col_w, ax_w, row_wh, col_wh, ax_wh);
  k_local<<<dim3(H_ / 4, B_), 256, 0, stream>>>(x, dw1_w, dw1_b, dw2_w, dw2_b, pw1_w, pw1_b, zsum);

  size_t sh_row = (size_t)64 * (C_ + 8) * 2 + 64 * 50 * 4 + 64 * 66 * 4 + 64 * 64 * 2;  // ~70 KB
  k_rowattn<<<dim3(H_, B_), 256, sh_row, stream>>>(x, row_wh, row_b, orow);

  size_t sh_col = (size_t)64 * 18 * 2 + 64 * 50 * 4 + 64 * 66 * 4 + 64 * 64 * 2 + 64 * 18 * 2 + 256 * 4;
  k_colattn<<<dim3(W_, B_), 256, sh_col, stream>>>(orow, col_wh, col_b, ax_wh, ax_b, axsum);

  k_fuse<<<1, 256, 0, stream>>>(gw, zsum, axsum, pw2_w, pw2_b, fusion_w, finalw);

  int n4 = B_ * C_ * HW / 4;
  k_apply<<<(n4 + 255) / 256, 256, 0, stream>>>((const v4f*)x, finalw, (v4f*)out, n4);
}
